// WindowAttention_34394098106738
// MI455X (gfx1250) — matrix operand order repaired and checked by reading, not compiled, not run on hardware
//
#include <hip/hip_runtime.h>

typedef __attribute__((ext_vector_type(16))) _Float16 v16h;
typedef __attribute__((ext_vector_type(8)))  _Float16 v8h;
typedef __attribute__((ext_vector_type(4)))  _Float16 v4h;
typedef __attribute__((ext_vector_type(8)))  float    v8f;

#define WMMA_F16(A, B, C) \
  __builtin_amdgcn_wmma_f32_16x16x32_f16(false, (A), false, (B), (short)0, (C), false, false)

// A-operand (16x32 f16, MxK): lane L holds row M=L%16; elements = K[kbase+16g .. +15]
// (g = L/16, g8 = 8*g)
__device__ __forceinline__ v16h ldA(const _Float16* row, int kbase, int g8) {
  return *(const v16h*)(row + kbase + 2 * g8);
}
// B-operand (32x16 f16, KxN): lane L holds col N=L%16; elements = K[kbase+16g .. +15]
__device__ __forceinline__ v16h ldB(const _Float16* row, int kbase, int g16) {
  return *(const v16h*)(row + kbase + g16);
}

// ---- prep: transpose+convert weights to f16, build 4 bias variants -------------
__global__ void prep_kernel(const float* __restrict__ w_qkv,
                            const float* __restrict__ w_out,
                            const float* __restrict__ pos,
                            const float* __restrict__ ul,
                            const float* __restrict__ lr,
                            _Float16* __restrict__ wqkvT,   // [384][128]
                            _Float16* __restrict__ woutT,   // [128][128]
                            float* __restrict__ biasv) {    // [4][64][64]
  int idx = blockIdx.x * 256 + threadIdx.x;
  if (idx < 384 * 128) {
    int c = idx >> 7, k = idx & 127;
    wqkvT[idx] = (_Float16)w_qkv[k * 384 + c];
  } else if (idx < 384 * 128 + 128 * 128) {
    int t = idx - 384 * 128;
    int c = t >> 7, k = t & 127;
    woutT[t] = (_Float16)w_out[k * 128 + c];
  } else if (idx < 384 * 128 + 128 * 128 + 4 * 64 * 64) {
    int t = idx - (384 * 128 + 128 * 128);
    int m = t >> 12, ij = t & 4095;
    int i = ij >> 6, j = ij & 63;
    int r0 = (i >> 3) - (j >> 3) + 7;
    int r1 = (i & 7) - (j & 7) + 7;
    float v = pos[r0 * 15 + r1];
    if (m & 1) v += ul[ij];
    if (m & 2) v += lr[ij];
    biasv[t] = v;
  }
}

// ---- fused shifted-window attention: one workgroup per (batch, window) ---------
__global__ __launch_bounds__(256)
void swin_attn_kernel(const float* __restrict__ x,
                      const _Float16* __restrict__ wqkvT,
                      const _Float16* __restrict__ woutT,
                      const float* __restrict__ b_out,
                      const float* __restrict__ biasv,
                      float* __restrict__ out) {
  extern __shared__ _Float16 smem[];
  _Float16* xo   = smem;                  // [64][128]  staged x, reused as o
  _Float16* qs   = smem + 1 * 64 * 128;   // [64][128]  (pre-scaled by 1/sqrt(32))
  _Float16* ks   = smem + 2 * 64 * 128;   // [64][128]
  _Float16* vTs  = smem + 3 * 64 * 128;   // [128][64]  v transposed
  _Float16* attn = smem + 4 * 64 * 128;   // [4][64][64]

  const int bb = blockIdx.x >> 8;
  const int w  = blockIdx.x & 255;
  const int wy = w >> 4, wx = w & 15;
  const int lane = threadIdx.x & 31;
  const int wv   = threadIdx.x >> 5;
  const int lrow = lane & 15;
  const int g8   = (lane >> 4) << 3;
  const int g16  = (lane >> 4) << 4;

  // Phase 0: stage rolled x window, fp32 -> f16 (roll -4,-4 folded into index)
  for (int v = threadIdx.x; v < 64 * 32; v += 256) {
    int tok = v >> 5;
    int c4  = (v & 31) << 2;
    int gy = (wy * 8 + (tok >> 3) + 4) & 127;
    int gx = (wx * 8 + (tok & 7) + 4) & 127;
    const float4 f =
        *(const float4*)(x + ((((size_t)bb * 128 + gy) * 128 + gx) << 7) + c4);
    v4h hv = { (_Float16)f.x, (_Float16)f.y, (_Float16)f.z, (_Float16)f.w };
    *(v4h*)(xo + tok * 128 + c4) = hv;
  }
  __syncthreads();

  // Phase 1: QKV projection (64x384, K=128).  Each wave owns 3 N-tiles and
  // holds B in registers across all 4 M-tiles -> weights read once per block.
  {
    const float scale = 0.17677669529663687f;  // 1/sqrt(32), folded into q
    const _Float16* aroot = xo + lrow * 128;
    for (int tn = 0; tn < 3; ++tn) {
      const int nt = __builtin_amdgcn_readfirstlane(wv * 3 + tn);  // 0..23
      const _Float16* brow = wqkvT + (nt * 16 + lrow) * 128;
      v16h Bt[4];
#pragma unroll
      for (int kk = 0; kk < 4; ++kk) Bt[kk] = ldB(brow, kk * 32, g16);

#pragma unroll
      for (int mt = 0; mt < 4; ++mt) {
        const _Float16* arow = aroot + mt * 16 * 128;
        v8f acc = {};
#pragma unroll
        for (int kk = 0; kk < 4; ++kk)
          acc = WMMA_F16(ldA(arow, kk * 32, g8), Bt[kk], acc);

        const int rb = mt * 16 + g8;
        if (nt < 8) {                     // q channels nt*16.. (pre-scaled)
          _Float16* dst = qs + rb * 128 + nt * 16 + lrow;
#pragma unroll
          for (int r = 0; r < 8; ++r) dst[r * 128] = (_Float16)(acc[r] * scale);
        } else if (nt < 16) {             // k channels (nt-8)*16..
          _Float16* dst = ks + rb * 128 + (nt - 8) * 16 + lrow;
#pragma unroll
          for (int r = 0; r < 8; ++r) dst[r * 128] = (_Float16)acc[r];
        } else {                          // v channel (nt-16)*16+lrow, transposed
          v8h hv;
#pragma unroll
          for (int r = 0; r < 8; ++r) hv[r] = (_Float16)acc[r];
          *(v8h*)(vTs + ((nt - 16) * 16 + lrow) * 64 + rb) = hv;
        }
      }
    }
  }
  __syncthreads();

  // Phase 2: attention; head h = wv/2, row-half = wv&1 (rows half*32 .. +31)
  {
    const int h = wv >> 1;
    const int half = wv & 1;

    v16h qa[2], kb[4];
#pragma unroll
    for (int mi = 0; mi < 2; ++mi)
      qa[mi] = ldA(qs + ((half * 2 + mi) * 16 + lrow) * 128 + h * 32, 0, g8);
#pragma unroll
    for (int nt = 0; nt < 4; ++nt)
      kb[nt] = ldB(ks + (nt * 16 + lrow) * 128 + h * 32, 0, g16);

    const int var = ((wy == 15) ? 1 : 0) | ((wx == 15) ? 2 : 0);
    const float* bv = biasv + var * 4096;

    // dots = (q*scale) . k + bias  (bias preloaded into the WMMA accumulator)
    v8f dts[2][4];
#pragma unroll
    for (int mi = 0; mi < 2; ++mi) {
      const int ib = (half * 2 + mi) * 16 + g8;
#pragma unroll
      for (int nt = 0; nt < 4; ++nt) {
        const float* bp = bv + ib * 64 + nt * 16 + lrow;
        v8f binit;
#pragma unroll
        for (int r = 0; r < 8; ++r) binit[r] = bp[r * 64];
        dts[mi][nt] = WMMA_F16(qa[mi], kb[nt], binit);
      }
    }

    _Float16* ah = attn + h * 64 * 64;
#pragma unroll
    for (int mi = 0; mi < 2; ++mi) {
      const int ib = (half * 2 + mi) * 16 + g8;
#pragma unroll
      for (int r = 0; r < 8; ++r) {
        float vals[4];
#pragma unroll
        for (int nt = 0; nt < 4; ++nt) vals[nt] = dts[mi][nt][r];
        float mx = fmaxf(fmaxf(vals[0], vals[1]), fmaxf(vals[2], vals[3]));
#pragma unroll
        for (int s = 1; s < 16; s <<= 1) mx = fmaxf(mx, __shfl_xor(mx, s, 32));
        float sum = 0.f;
#pragma unroll
        for (int nt = 0; nt < 4; ++nt) {
          vals[nt] = __expf(vals[nt] - mx);
          sum += vals[nt];
        }
#pragma unroll
        for (int s = 1; s < 16; s <<= 1) sum += __shfl_xor(sum, s, 32);
        float inv = 1.0f / sum;
        _Float16* arow = ah + (ib + r) * 64 + lrow;
#pragma unroll
        for (int nt = 0; nt < 4; ++nt)
          arow[nt * 16] = (_Float16)(vals[nt] * inv);
      }
    }

    // out = attn @ v  (K=64 -> two WMMA k-steps)
    v16h aa[2][2], vb[2][2];
#pragma unroll
    for (int mi = 0; mi < 2; ++mi) {
      const _Float16* ar = ah + ((half * 2 + mi) * 16 + lrow) * 64;
#pragma unroll
      for (int kk = 0; kk < 2; ++kk) aa[mi][kk] = ldA(ar, kk * 32, g8);
    }
#pragma unroll
    for (int nd = 0; nd < 2; ++nd) {
      const _Float16* br = vTs + (h * 32 + nd * 16 + lrow) * 64;
#pragma unroll
      for (int kk = 0; kk < 2; ++kk) vb[nd][kk] = ldB(br, kk * 32, g16);
    }
#pragma unroll
    for (int mi = 0; mi < 2; ++mi)
#pragma unroll
      for (int nd = 0; nd < 2; ++nd) {
        v8f o = {};
        o = WMMA_F16(aa[mi][0], vb[nd][0], o);
        o = WMMA_F16(aa[mi][1], vb[nd][1], o);
        const int c = h * 32 + nd * 16 + lrow;  // merged channel = h*32 + d
        _Float16* dst = xo + ((half * 2 + mi) * 16 + g8) * 128 + c;
#pragma unroll
        for (int r = 0; r < 8; ++r) dst[r * 128] = (_Float16)o[r];
      }
  }
  __syncthreads();

  // Phase 3: output projection + bias.  Each wave owns one N-tile (B held in
  // registers) and iterates the 4 M-tiles -> woutT read once per block.
  {
    const _Float16* brow = woutT + (wv * 16 + lrow) * 128;
    v16h Bt[4];
#pragma unroll
    for (int kk = 0; kk < 4; ++kk) Bt[kk] = ldB(brow, kk * 32, g16);
    const int c = wv * 16 + lrow;
    const float bias = b_out[c];
    const int gxb = wx * 8 + 4;

#pragma unroll
    for (int mt = 0; mt < 4; ++mt) {
      const _Float16* arow = xo + (mt * 16 + lrow) * 128;
      v8f acc = {};
#pragma unroll
      for (int kk = 0; kk < 4; ++kk)
        acc = WMMA_F16(ldA(arow, kk * 32, g8), Bt[kk], acc);

      const int rb = mt * 16 + g8;                    // rb % 8 == 0
      const int gy = (wy * 8 + (rb >> 3) + 4) & 127;  // constant over r
      float* obase = out + (((size_t)bb * 128 + gy) << 14) + c;
#pragma unroll
      for (int r = 0; r < 8; ++r)
        obase[((gxb + r) & 127) << 7] = acc[r] + bias;
    }
  }
}

extern "C" void kernel_launch(void* const* d_in, const int* in_sizes, int n_in,
                              void* d_out, int out_size, void* d_ws, size_t ws_size,
                              hipStream_t stream) {
  const float* x     = (const float*)d_in[0];
  const float* w_qkv = (const float*)d_in[1];
  const float* w_out = (const float*)d_in[2];
  const float* b_out = (const float*)d_in[3];
  const float* pos   = (const float*)d_in[4];
  const float* ul    = (const float*)d_in[5];
  const float* lr    = (const float*)d_in[6];

  _Float16* wqkvT = (_Float16*)d_ws;             // 384*128 f16 = 96 KB
  _Float16* woutT = wqkvT + 384 * 128;           // 128*128 f16 = 32 KB
  float*    biasv = (float*)(woutT + 128 * 128); // 4*64*64 f32 = 64 KB

  prep_kernel<<<320, 256, 0, stream>>>(w_qkv, w_out, pos, ul, lr,
                                       wqkvT, woutT, biasv);

  int B = in_sizes[0] / (128 * 128 * 128);
  size_t lds = (size_t)(4 * 64 * 128 + 4 * 64 * 64) * sizeof(_Float16); // 96 KB
  swin_attn_kernel<<<B * 256, 256, lds, stream>>>(x, wqkvT, woutT, b_out,
                                                  biasv, (float*)d_out);
}
